// SpecAttn_90005334655734
// MI455X (gfx1250) — compile-verified, matrix operand order checked by reading, not run on hardware
//
#include <hip/hip_runtime.h>

// ---------------------------------------------------------------------------
// CDNA5 (gfx1250) fused attention pipeline, wave32 + v_wmma_f32_16x16x32_f16
// v2: software-pipelined global->LDS staging (register prefetch of next K
//     tile, loads issued before the WMMA burst), padded LDS rows (40 halves =
//     80B: conflict-free ds_load_b128 fragment reads), global_prefetch hints.
// ---------------------------------------------------------------------------

typedef __attribute__((ext_vector_type(16))) _Float16 v16h;
typedef __attribute__((ext_vector_type(8)))  _Float16 v8h;
typedef __attribute__((ext_vector_type(4)))  _Float16 v4h;
typedef __attribute__((ext_vector_type(8)))  float    v8f;

#define TILE_M 128
#define TILE_N 128
#define TILE_K 32
#define LDK    40   // padded LDS row stride in halves (80B, 16B aligned)

// Gather one 16x32 f16 WMMA operand fragment from an LDS tile stored with a
// contiguous 32-half K-row per matrix row (row stride LDK halves).
// Per ISA 16-bit A/B layout:
//   lanes 0-15  hold K = 0..7  and 16..23
//   lanes 16-31 hold K = 8..15 and 24..31
// -> two 16-byte LDS reads (ds_load_b128) per fragment.
__device__ __forceinline__ v16h load_frag(const _Float16* row32, int lane) {
    const int kb = (lane & 16) ? 8 : 0;
    v8h lo = *(const v8h*)(row32 + kb);
    v8h hi = *(const v8h*)(row32 + kb + 16);
    v16h r;
#pragma unroll
    for (int i = 0; i < 8; ++i) { r[i] = lo[i]; r[i + 8] = hi[i]; }
    return r;
}

// Generic tiled WMMA GEMM:  C = scale * (A @ B) [+ bias]
//   A_F32    : A is f32 in memory (convert to f16 at load), else f16.
//   B_LAYOUT : 0 = B f32 [K][N] (convert + transpose into LDS [N][K])
//              1 = B f16 [N][K] (direct row copy into LDS)
//              2 = B f16 [K][N] (transpose into LDS [N][K])
//   OUT_MODE : 0 = f16 out (scale only), 1 = f16 out + f32 bias,
//              2 = f32 out + f32 bias
// Block = 128 threads = 4 wave32s; each wave computes a 64x64 tile as a 4x4
// grid of 16x16 f32 accumulators; block tile 128x128; K-step 32.
// All dims are multiples of 128/32 in this problem, so no bounds checks.
template <bool A_F32, int B_LAYOUT, int OUT_MODE>
__global__ __launch_bounds__(128)
void gemm_wmma(const void* __restrict__ Ap, const void* __restrict__ Bp,
               const float* __restrict__ bias, void* __restrict__ Cp,
               int M, int N, int K, int lda, int ldb, int ldc, float scale,
               long long sAbatch, long long sBbatch, long long sCbatch) {
    (void)M; (void)N;
    __shared__ _Float16 ldsA[TILE_M * LDK];
    __shared__ _Float16 ldsB[TILE_N * LDK];

    const int tid   = threadIdx.x;
    const int lane  = tid & 31;
    const int wid   = tid >> 5;     // 0..3
    const int warpM = wid >> 1;     // 0..1
    const int warpN = wid & 1;      // 0..1

    const int m0 = blockIdx.y * TILE_M;
    const int n0 = blockIdx.x * TILE_N;
    const long long z = blockIdx.z;

    const float*    Af = (const float*)Ap    + z * sAbatch;
    const _Float16* Ah = (const _Float16*)Ap + z * sAbatch;
    const float*    Bf = (const float*)Bp    + z * sBbatch;
    const _Float16* Bh = (const _Float16*)Bp + z * sBbatch;
    float*          Cf = (float*)Cp          + z * sCbatch;
    _Float16*       Ch = (_Float16*)Cp       + z * sCbatch;

    // Register staging for the software pipeline (converted to f16 at load so
    // the live range is half the size of raw f32 staging).
    v4h arF[8]; v8h arH[4];     // A tile chunks (only one set used per inst.)
    v4h brF[8]; v8h brH[4];     // B tile chunks

    auto loadA = [&](int k0) {
        if constexpr (A_F32) {
#pragma unroll
            for (int i = 0; i < 8; ++i) {           // 128x32 f32, float4 chunks
                const int idx = tid + i * 128;
                const int r = idx >> 3, kg = (idx & 7) << 2;
                const float4 f =
                    *(const float4*)(Af + (size_t)(m0 + r) * lda + k0 + kg);
                arF[i] = v4h{(_Float16)f.x, (_Float16)f.y,
                             (_Float16)f.z, (_Float16)f.w};
            }
        } else {
#pragma unroll
            for (int i = 0; i < 4; ++i) {           // 128x32 f16, v8h chunks
                const int idx = tid + i * 128;
                const int r = idx >> 2, kg = (idx & 3) << 3;
                arH[i] = *(const v8h*)(Ah + (size_t)(m0 + r) * lda + k0 + kg);
            }
        }
    };
    auto storeA = [&]() {
        if constexpr (A_F32) {
#pragma unroll
            for (int i = 0; i < 8; ++i) {
                const int idx = tid + i * 128;
                const int r = idx >> 3, kg = (idx & 7) << 2;
                *(v4h*)&ldsA[r * LDK + kg] = arF[i];     // ds_store_b64
            }
        } else {
#pragma unroll
            for (int i = 0; i < 4; ++i) {
                const int idx = tid + i * 128;
                const int r = idx >> 2, kg = (idx & 3) << 3;
                *(v8h*)&ldsA[r * LDK + kg] = arH[i];     // ds_store_b128
            }
        }
    };
    auto loadB = [&](int k0) {
        if constexpr (B_LAYOUT == 0) {
#pragma unroll
            for (int i = 0; i < 8; ++i) {           // f32 [K][N], float4 over N
                const int idx = tid + i * 128;
                const int kk = idx >> 5, ng = (idx & 31) << 2;
                const float4 f =
                    *(const float4*)(Bf + (size_t)(k0 + kk) * ldb + n0 + ng);
                brF[i] = v4h{(_Float16)f.x, (_Float16)f.y,
                             (_Float16)f.z, (_Float16)f.w};
            }
        } else if constexpr (B_LAYOUT == 1) {
#pragma unroll
            for (int i = 0; i < 4; ++i) {           // f16 [N][K], direct rows
                const int idx = tid + i * 128;
                const int n = idx >> 2, kg = (idx & 3) << 3;
                brH[i] = *(const v8h*)(Bh + (size_t)(n0 + n) * ldb + k0 + kg);
            }
        } else {
#pragma unroll
            for (int i = 0; i < 4; ++i) {           // f16 [K][N], v8h over N
                const int idx = tid + i * 128;
                const int kk = idx >> 4, ng = (idx & 15) << 3;
                brH[i] = *(const v8h*)(Bh + (size_t)(k0 + kk) * ldb + n0 + ng);
            }
        }
    };
    auto storeB = [&]() {
        if constexpr (B_LAYOUT == 0) {
#pragma unroll
            for (int i = 0; i < 8; ++i) {           // transpose-scatter
                const int idx = tid + i * 128;
                const int kk = idx >> 5, ng = (idx & 31) << 2;
#pragma unroll
                for (int j = 0; j < 4; ++j)
                    ldsB[(ng + j) * LDK + kk] = brF[i][j];
            }
        } else if constexpr (B_LAYOUT == 1) {
#pragma unroll
            for (int i = 0; i < 4; ++i) {
                const int idx = tid + i * 128;
                const int n = idx >> 2, kg = (idx & 3) << 3;
                *(v8h*)&ldsB[n * LDK + kg] = brH[i];     // ds_store_b128
            }
        } else {
#pragma unroll
            for (int i = 0; i < 4; ++i) {           // transpose-scatter
                const int idx = tid + i * 128;
                const int kk = idx >> 4, ng = (idx & 15) << 3;
#pragma unroll
                for (int j = 0; j < 8; ++j)
                    ldsB[(ng + j) * LDK + kk] = brH[i][j];
            }
        }
    };

    v8f acc[4][4];
#pragma unroll
    for (int mt = 0; mt < 4; ++mt)
#pragma unroll
        for (int nt = 0; nt < 4; ++nt)
#pragma unroll
            for (int r = 0; r < 8; ++r) acc[mt][nt][r] = 0.0f;

    loadA(0);
    loadB(0);
    for (int k0 = 0; k0 < K; k0 += TILE_K) {
        if (k0) __syncthreads();     // all waves done reading previous tile
        storeA();
        storeB();
        __syncthreads();             // staging visible to all waves

        const int kn = k0 + TILE_K;
        if (kn < K) {                // issue next tile's global loads *before*
            loadA(kn);               // the WMMA burst -> latency overlap
            loadB(kn);
        }
        if (kn + TILE_K < K) {       // warm L2 two tiles ahead
            const int k2 = kn + TILE_K;
            if constexpr (A_F32)
                __builtin_prefetch(Af + (size_t)(m0 + (tid >> 3)) * lda + k2, 0, 2);
            else
                __builtin_prefetch(Ah + (size_t)(m0 + (tid >> 2)) * lda + k2, 0, 2);
            if constexpr (B_LAYOUT == 0)
                __builtin_prefetch(Bf + (size_t)(k2 + (tid >> 5)) * ldb + n0 +
                                       ((tid & 31) << 2), 0, 2);
            else if constexpr (B_LAYOUT == 1)
                __builtin_prefetch(Bh + (size_t)(n0 + (tid >> 2)) * ldb + k2, 0, 2);
            else
                __builtin_prefetch(Bh + (size_t)(k2 + (tid >> 4)) * ldb + n0 +
                                       ((tid & 15) << 3), 0, 2);
        }

        v16h afrag[4], bfrag[4];
#pragma unroll
        for (int mt = 0; mt < 4; ++mt)
            afrag[mt] = load_frag(
                &ldsA[(warpM * 64 + mt * 16 + (lane & 15)) * LDK], lane);
#pragma unroll
        for (int nt = 0; nt < 4; ++nt)
            bfrag[nt] = load_frag(
                &ldsB[(warpN * 64 + nt * 16 + (lane & 15)) * LDK], lane);

#pragma unroll
        for (int mt = 0; mt < 4; ++mt)
#pragma unroll
            for (int nt = 0; nt < 4; ++nt)
                acc[mt][nt] = __builtin_amdgcn_wmma_f32_16x16x32_f16(
                    false, afrag[mt], false, bfrag[nt], (short)0,
                    acc[mt][nt], false, false);
    }

    // ---- epilogue: C/D layout => lane holds N=lane&15, VGPR r -> M row ----
    const int mbase = m0 + warpM * 64 + ((lane & 16) ? 8 : 0);
    const int nbase = n0 + warpN * 64 + (lane & 15);
#pragma unroll
    for (int mt = 0; mt < 4; ++mt) {
#pragma unroll
        for (int nt = 0; nt < 4; ++nt) {
            const int n = nbase + nt * 16;
            float bval = 0.0f;
            if constexpr (OUT_MODE >= 1) bval = bias[n];
#pragma unroll
            for (int r = 0; r < 8; ++r) {
                const int m = mbase + mt * 16 + r;
                const float v = acc[mt][nt][r] * scale + bval;
                if constexpr (OUT_MODE == 2)
                    Cf[(size_t)m * ldc + n] = v;
                else
                    Ch[(size_t)m * ldc + n] = (_Float16)v;
            }
        }
    }
}

// In-place row softmax over f16 energies (f32 math, v_exp_f32).
// One 256-thread block per row; writes normalized probabilities.
__global__ __launch_bounds__(256)
void softmax_inplace(_Float16* __restrict__ p, int ncols) {
    const size_t base = (size_t)blockIdx.x * (size_t)ncols;
    const int tid = threadIdx.x;
    __shared__ float red[256];

    float m = -3.0e38f;
    for (int c = tid; c < ncols; c += 256)
        m = fmaxf(m, (float)p[base + c]);
    red[tid] = m;
    __syncthreads();
    for (int s = 128; s > 0; s >>= 1) {
        if (tid < s) red[tid] = fmaxf(red[tid], red[tid + s]);
        __syncthreads();
    }
    m = red[0];
    __syncthreads();

    float l = 0.0f;
    for (int c = tid; c < ncols; c += 256)
        l += __expf((float)p[base + c] - m);
    red[tid] = l;
    __syncthreads();
    for (int s = 128; s > 0; s >>= 1) {
        if (tid < s) red[tid] += red[tid + s];
        __syncthreads();
    }
    const float inv = 1.0f / red[0];

    for (int c = tid; c < ncols; c += 256)
        p[base + c] = (_Float16)(__expf((float)p[base + c] - m) * inv);
}

extern "C" void kernel_launch(void* const* d_in, const int* in_sizes, int n_in,
                              void* d_out, int out_size, void* d_ws,
                              size_t ws_size, hipStream_t stream) {
    (void)in_sizes; (void)n_in; (void)out_size; (void)ws_size;

    const float* x  = (const float*)d_in[0];
    const float* wq = (const float*)d_in[1];
    const float* bq = (const float*)d_in[2];
    const float* wk = (const float*)d_in[3];
    const float* bk = (const float*)d_in[4];
    const float* wv = (const float*)d_in[5];
    const float* bv = (const float*)d_in[6];
    const float* wo = (const float*)d_in[7];
    const float* bo = (const float*)d_in[8];

    constexpr int       Bb = 8, S = 2048, H = 1024;
    constexpr long long MS = (long long)Bb * S;  // 16384 rows

    // Workspace layout (bytes):
    //   q    : 16384*1024 f16 = 32 MB   @ 0
    //   k    : 32 MB                    @ 32 MB
    //   v    : 32 MB                    @ 64 MB
    //   E/P  : 8*2048*2048 f16 = 64 MB  @ 96 MB   (softmax runs in place)
    //   ctx  : 32 MB                    @ 160 MB
    char* ws = (char*)d_ws;
    _Float16* qb = (_Float16*)(ws);
    _Float16* kb = (_Float16*)(ws + (size_t)33554432);
    _Float16* vb = (_Float16*)(ws + (size_t)2 * 33554432);
    _Float16* eb = (_Float16*)(ws + (size_t)3 * 33554432);
    _Float16* cb = (_Float16*)(ws + (size_t)3 * 33554432 + 67108864);

    const dim3 blk(128);

    // 1) QKV projections: [16384,1024] = x @ W + b   (f32 in, f16 out)
    const dim3 gqkv(H / TILE_N, MS / TILE_M, 1);
    gemm_wmma<true, 0, 1><<<gqkv, blk, 0, stream>>>(
        x, wq, bq, qb, (int)MS, H, H, H, H, H, 1.0f, 0, 0, 0);
    gemm_wmma<true, 0, 1><<<gqkv, blk, 0, stream>>>(
        x, wk, bk, kb, (int)MS, H, H, H, H, H, 1.0f, 0, 0, 0);
    gemm_wmma<true, 0, 1><<<gqkv, blk, 0, stream>>>(
        x, wv, bv, vb, (int)MS, H, H, H, H, H, 1.0f, 0, 0, 0);

    // 2) energy = q @ k^T / sqrt(H)  per batch; k rows are K-contiguous B
    const dim3 ge(S / TILE_N, S / TILE_M, Bb);
    gemm_wmma<false, 1, 0><<<ge, blk, 0, stream>>>(
        qb, kb, nullptr, eb, S, S, H, H, H, S, 0.03125f,
        (long long)S * H, (long long)S * H, (long long)S * S);

    // 3) softmax rows in place (8*2048 rows of 2048)
    softmax_inplace<<<Bb * S, 256, 0, stream>>>(eb, S);

    // 4) ctx = P @ v  per batch (v is [kpos][h] -> transpose at staging)
    const dim3 gc(H / TILE_N, S / TILE_M, Bb);
    gemm_wmma<false, 2, 0><<<gc, blk, 0, stream>>>(
        eb, vb, nullptr, cb, S, H, S, S, H, H, 1.0f,
        (long long)S * S, (long long)S * H, (long long)S * H);

    // 5) out = ctx @ wo + bo   (f16 A, f32 W, f32 out)
    const dim3 go(H / TILE_N, MS / TILE_M, 1);
    gemm_wmma<false, 0, 2><<<go, blk, 0, stream>>>(
        cb, wo, bo, d_out, (int)MS, H, H, H, H, H, 1.0f, 0, 0, 0);
}
